// TransformerEncoder_32633161515614
// MI455X (gfx1250) — matrix operand order repaired and checked by reading, not compiled, not run on hardware
//
#include <hip/hip_runtime.h>
#include <hip/hip_bf16.h>
#include <cstdint>
#include <cstddef>

// ---------------- problem constants (match reference) ----------------
#define LAY 6
#define NH 16
#define DMODEL 1024
#define DHEAD 64
#define DFF 4096
#define BB 4
#define SS 1024
#define NEG_INF_F (-1.0e8f)

typedef __bf16 bf16_t;
typedef bf16_t v16bf __attribute__((ext_vector_type(16)));
typedef float  v8f   __attribute__((ext_vector_type(8)));
typedef int    v4i   __attribute__((ext_vector_type(4)));

union Frag {
    v16bf v;
    uint4 q[2];
};

// native bf16 convert (gfx1250 has hardware cvt; RNE)
static __device__ __forceinline__ bf16_t f2bf(float f) {
    return (bf16_t)f;
}

// pack two f32 -> one dword of 2 bf16 {lo, hi}
static __device__ __forceinline__ uint32_t pk2bf(float lo, float hi) {
#if __has_builtin(__builtin_amdgcn_cvt_pk_bf16_f32)
    auto v = __builtin_amdgcn_cvt_pk_bf16_f32(lo, hi);
    uint32_t u;
    __builtin_memcpy(&u, &v, 4);
    return u;
#else
    union { bf16_t b[2]; uint32_t u; } o;
    o.b[0] = f2bf(lo);
    o.b[1] = f2bf(hi);
    return o.u;
#endif
}

static __device__ __forceinline__ uint32_t pk2u16(uint16_t lo, uint16_t hi) {
    return (uint32_t)lo | ((uint32_t)hi << 16);
}

static __device__ __forceinline__ v8f wmma_bf16(const v16bf& a, const v16bf& b, const v8f& c) {
    return __builtin_amdgcn_wmma_f32_16x16x32_bf16(false, a, false, b, (short)0, c, false, false);
}

// ---- CDNA5 async global->LDS copy path (signature: int4 AS1*, int4 AS3*, imm, imm) ----
#if __has_builtin(__builtin_amdgcn_global_load_async_to_lds_b128)
#define ASYNC_LDS 1
typedef __attribute__((address_space(1))) v4i gv4i;
typedef __attribute__((address_space(3))) v4i lv4i;
static __device__ __forceinline__ void async_cp16(const void* g, void* l) {
    __builtin_amdgcn_global_load_async_to_lds_b128((gv4i*)g, (lv4i*)l, 0, 0);
}
static __device__ __forceinline__ void async_wait0() {
#if __has_builtin(__builtin_amdgcn_s_wait_asynccnt)
    __builtin_amdgcn_s_wait_asynccnt(0);
#else
    __asm__ volatile("s_wait_asynccnt 0" ::: "memory");
#endif
}
#endif

// =====================================================================
// Positional encoding: xo = x + sincos, xb = bf16(xo)
// =====================================================================
__global__ __launch_bounds__(256) void posenc_kernel(const float* __restrict__ x,
                                                     float* __restrict__ xo,
                                                     bf16_t* __restrict__ xb) {
    size_t i = (size_t)blockIdx.x * 256 + threadIdx.x;
    int d = (int)(i % DMODEL);
    int s = (int)((i / DMODEL) % SS);
    int t = d & 511;                                    // d or d-512
    float inv = __expf(-(float)t * 0.018024149455f);    // ln(1e4)/511
    float a = (float)s * inv;
    float sig = (d < 512) ? __sinf(a) : __cosf(a);
    float o = x[i] + sig;
    xo[i] = o;
    xb[i] = f2bf(o);
}

// =====================================================================
// GEMM: C[M,N] = act(scale*(A_bf16[M,K] @ W_f32[K,N] + bias)) -> f32/bf16
// block tile 128x128, 8 waves (2x4) of 64x32, K-step 32, WMMA bf16
// =====================================================================
#define BM 128
#define BN 128
#define BK 32

__global__ __launch_bounds__(256) void gemm_kernel(const bf16_t* __restrict__ A,
                                                   const float* __restrict__ W,
                                                   const float* __restrict__ bias,
                                                   float* __restrict__ Cf,
                                                   bf16_t* __restrict__ Cb,
                                                   int M, int N, int K,
                                                   float scale, int relu) {
    __shared__ bf16_t sA[BM * BK];      // [row][k]
    __shared__ bf16_t sBt[BN * BK];     // [n][k]   (transposed B)

    const int tid  = threadIdx.x;
    const int lane = tid & 31;
    const int wv   = tid >> 5;          // 0..7
    const int wM   = wv >> 2;           // 0..1
    const int wN   = wv & 3;            // 0..3
    const int half = lane >> 4;
    const int l16  = lane & 15;
    const int bM = blockIdx.y * BM;
    const int bN = blockIdx.x * BN;

    v8f acc[4][2] = {};

    for (int k0 = 0; k0 < K; k0 += BK) {
        // ---- stage A tile (bf16, row-major): 2x16B per thread ----
        {
            int row = tid >> 1;               // 0..127
            int seg = (tid & 1) * 16;         // element offset 0 / 16
            const uint4* g = (const uint4*)(A + (size_t)(bM + row) * K + k0 + seg);
            bf16_t* l0 = &sA[row * BK + seg];
#ifdef ASYNC_LDS
            async_cp16(g,     l0);
            async_cp16(g + 1, l0 + 8);
#else
            uint4 d0 = g[0];
            uint4 d1 = g[1];
            *(uint4*)l0       = d0;
            *(uint4*)(l0 + 8) = d1;
#endif
        }
        // ---- stage B tile transposed: two K-rows packed per dword ----
        {
            int kp   = tid >> 4;              // 0..15 -> k = 2*kp
            int nseg = (tid & 15) * 8;        // 0..120
            const float* r0 = W + (size_t)(k0 + 2 * kp) * N + bN + nseg;
            const float* r1 = r0 + N;
            float4 a0 = ((const float4*)r0)[0];
            float4 a1 = ((const float4*)r0)[1];
            float4 b0 = ((const float4*)r1)[0];
            float4 b1 = ((const float4*)r1)[1];
            uint32_t* dst0 = (uint32_t*)&sBt[(nseg + 0) * BK + 2 * kp];
            // stride between consecutive n rows: BK bf16 = BK/2 dwords
            const int strd = BK / 2;
            dst0[0 * strd] = pk2bf(a0.x, b0.x);
            dst0[1 * strd] = pk2bf(a0.y, b0.y);
            dst0[2 * strd] = pk2bf(a0.z, b0.z);
            dst0[3 * strd] = pk2bf(a0.w, b0.w);
            dst0[4 * strd] = pk2bf(a1.x, b1.x);
            dst0[5 * strd] = pk2bf(a1.y, b1.y);
            dst0[6 * strd] = pk2bf(a1.z, b1.z);
            dst0[7 * strd] = pk2bf(a1.w, b1.w);
        }
#ifdef ASYNC_LDS
        async_wait0();
#endif
        __syncthreads();

        if (k0 + BK < K)   // prefetch next weight K-slice
            __builtin_prefetch(W + (size_t)(k0 + BK) * N + bN + (tid & 7) * 16, 0, 0);

        // ---- B fragments (32x16: lane half -> K 8h..8h+7, 16+8h..16+8h+7) ----
        Frag bf[2];
#pragma unroll
        for (int n = 0; n < 2; ++n) {
            const bf16_t* p = &sBt[(wN * 32 + n * 16 + l16) * BK + half * 16];
            bf[n].q[0] = *(const uint4*)(p - half * 8);
            bf[n].q[1] = *(const uint4*)(p + 16 - half * 8);
        }
        // ---- A fragments + WMMA ----
#pragma unroll
        for (int m = 0; m < 4; ++m) {
            Frag af;
            const bf16_t* p = &sA[(wM * 64 + m * 16 + l16) * BK];
            af.q[0] = *(const uint4*)(p + half * 8);
            af.q[1] = *(const uint4*)(p + 16 + half * 8);
#pragma unroll
            for (int n = 0; n < 2; ++n)
                acc[m][n] = wmma_bf16(af.v, bf[n].v, acc[m][n]);
        }
        __syncthreads();
    }

    // ---- epilogue: bias, scale, relu, dual store ----
#pragma unroll
    for (int m = 0; m < 4; ++m) {
#pragma unroll
        for (int n = 0; n < 2; ++n) {
            int col = bN + wN * 32 + n * 16 + l16;
            float bv = bias ? bias[col] : 0.0f;
#pragma unroll
            for (int r = 0; r < 8; ++r) {
                int row = bM + wM * 64 + m * 16 + r + half * 8;
                float v = (acc[m][n][r] + bv) * scale;
                if (relu) v = fmaxf(v, 0.0f);
                size_t idx = (size_t)row * N + col;
                if (Cf) Cf[idx] = v;
                if (Cb) Cb[idx] = f2bf(v);
            }
        }
    }
}

// =====================================================================
// Flash attention: block = 4 waves x 32 query rows; loop 64-key blocks.
// =====================================================================
__global__ __launch_bounds__(128) void attn_kernel(const bf16_t* __restrict__ Q,
                                                   const bf16_t* __restrict__ Kb,
                                                   const bf16_t* __restrict__ Vb,
                                                   const float* __restrict__ mask,
                                                   bf16_t* __restrict__ O) {
    const int bh = blockIdx.y;
    const int b = bh >> 4;              // NH = 16
    const int h = bh & 15;
    const int tid  = threadIdx.x;
    const int lane = tid & 31;
    const int wv   = tid >> 5;          // 0..3
    const int half = lane >> 4;
    const int l16  = lane & 15;
    const int qrow0 = blockIdx.x * 128 + wv * 32;

    __shared__ bf16_t vt[64 * 64];          // [dh][key]
    __shared__ bf16_t pl[4][32 * 64];       // per-wave P tile

    // ---- load Q fragments [m=2][kstep=2] ----
    Frag qa[2][2];
#pragma unroll
    for (int m = 0; m < 2; ++m)
#pragma unroll
        for (int ks = 0; ks < 2; ++ks) {
            int row = qrow0 + m * 16 + l16;
            const bf16_t* p = Q + ((size_t)(b * SS + row)) * DMODEL + h * 64 + ks * 32;
            qa[m][ks].q[0] = *(const uint4*)(p + half * 8);
            qa[m][ks].q[1] = *(const uint4*)(p + 16 + half * 8);
        }

    v8f out[2][4] = {};
    float rmax[2][8], rsum[2][8];
#pragma unroll
    for (int m = 0; m < 2; ++m)
#pragma unroll
        for (int r = 0; r < 8; ++r) { rmax[m][r] = -1.0e30f; rsum[m][r] = 0.0f; }

    for (int kb = 0; kb < SS / 64; ++kb) {
        const int kbase = kb * 64;

        // ---- stage V transposed: vt[dh][key], two keys packed per dword ----
        {
            int kp    = tid & 31;            // keys 2kp, 2kp+1
            int dhseg = (tid >> 5) * 16;     // 0,16,32,48
            const bf16_t* g0 = Vb + ((size_t)(b * SS + kbase + 2 * kp)) * DMODEL + h * 64 + dhseg;
            const bf16_t* g1 = g0 + DMODEL;
            union { uint4 u; uint16_t e[8]; } x0, x1, y0, y1;
            x0.u = ((const uint4*)g0)[0];
            x1.u = ((const uint4*)g0)[1];
            y0.u = ((const uint4*)g1)[0];
            y1.u = ((const uint4*)g1)[1];
            uint32_t* dst = (uint32_t*)&vt[dhseg * 64 + 2 * kp];
            const int strd = 32;             // 64 bf16 per dh-row = 32 dwords
#pragma unroll
            for (int j = 0; j < 8; ++j)
                dst[j * strd] = pk2u16(x0.e[j], y0.e[j]);
#pragma unroll
            for (int j = 0; j < 8; ++j)
                dst[(8 + j) * strd] = pk2u16(x1.e[j], y1.e[j]);
        }
        __syncthreads();

        // ---- logits S = Q @ K^T  (K row-major == transposed-B layout) ----
        v8f s[2][4] = {};
#pragma unroll
        for (int n = 0; n < 4; ++n) {
#pragma unroll
            for (int ks = 0; ks < 2; ++ks) {
                Frag kf;
                int col = kbase + n * 16 + l16;
                const bf16_t* p = Kb + ((size_t)(b * SS + col)) * DMODEL + h * 64 + ks * 32 + half * 16;
                kf.q[0] = *(const uint4*)(p - half * 8);
                kf.q[1] = *(const uint4*)(p + 16 - half * 8);
#pragma unroll
                for (int m = 0; m < 2; ++m)
                    s[m][n] = wmma_bf16(qa[m][ks].v, kf.v, s[m][n]);
            }
        }
        // ---- additive mask ----
#pragma unroll
        for (int n = 0; n < 4; ++n) {
            float am = (1.0f - mask[(size_t)b * SS + kbase + n * 16 + l16]) * NEG_INF_F;
#pragma unroll
            for (int m = 0; m < 2; ++m)
#pragma unroll
                for (int r = 0; r < 8; ++r) s[m][n][r] += am;
        }
        // ---- online softmax (row reductions across 16-lane group) ----
#pragma unroll
        for (int m = 0; m < 2; ++m) {
#pragma unroll
            for (int r = 0; r < 8; ++r) {
                float vmx = -1.0e30f;
#pragma unroll
                for (int n = 0; n < 4; ++n) vmx = fmaxf(vmx, s[m][n][r]);
#pragma unroll
                for (int off = 1; off < 16; off <<= 1)
                    vmx = fmaxf(vmx, __shfl_xor(vmx, off, 32));
                float mo = rmax[m][r];
                float mn = fmaxf(mo, vmx);
                float scl = __expf(mo - mn);
                float rs = 0.0f;
#pragma unroll
                for (int n = 0; n < 4; ++n) {
                    float p = __expf(s[m][n][r] - mn);
                    s[m][n][r] = p;
                    rs += p;
                }
#pragma unroll
                for (int off = 1; off < 16; off <<= 1)
                    rs += __shfl_xor(rs, off, 32);
                rsum[m][r] = rsum[m][r] * scl + rs;
                rmax[m][r] = mn;
#pragma unroll
                for (int n = 0; n < 4; ++n) out[m][n][r] *= scl;
            }
        }
        // ---- spill P (C layout) to LDS, re-read as A fragments ----
        bf16_t* myp = pl[wv];
#pragma unroll
        for (int m = 0; m < 2; ++m)
#pragma unroll
            for (int n = 0; n < 4; ++n)
#pragma unroll
                for (int r = 0; r < 8; ++r)
                    myp[(m * 16 + r + half * 8) * 64 + n * 16 + l16] = f2bf(s[m][n][r]);
        __syncthreads();

        Frag pa[2][2];
#pragma unroll
        for (int m = 0; m < 2; ++m)
#pragma unroll
            for (int ks = 0; ks < 2; ++ks) {
                const bf16_t* pp = myp + (m * 16 + l16) * 64 + ks * 32;
                pa[m][ks].q[0] = *(const uint4*)(pp + half * 8);
                pa[m][ks].q[1] = *(const uint4*)(pp + 16 + half * 8);
            }
        // ---- out += P @ V ----
#pragma unroll
        for (int n = 0; n < 4; ++n) {
            Frag vf[2];
#pragma unroll
            for (int ks = 0; ks < 2; ++ks) {
                const bf16_t* p = &vt[(n * 16 + l16) * 64 + ks * 32 + half * 16];
                vf[ks].q[0] = *(const uint4*)(p - half * 8);
                vf[ks].q[1] = *(const uint4*)(p + 16 - half * 8);
            }
#pragma unroll
            for (int m = 0; m < 2; ++m)
#pragma unroll
                for (int ks = 0; ks < 2; ++ks)
                    out[m][n] = wmma_bf16(pa[m][ks].v, vf[ks].v, out[m][n]);
        }
        __syncthreads();
    }

    // ---- normalize + store bf16 ----
#pragma unroll
    for (int m = 0; m < 2; ++m)
#pragma unroll
        for (int n = 0; n < 4; ++n) {
            int dh = n * 16 + l16;
#pragma unroll
            for (int r = 0; r < 8; ++r) {
                int row = qrow0 + m * 16 + r + half * 8;
                float v = out[m][n][r] / rsum[m][r];
                O[((size_t)(b * SS + row)) * DMODEL + h * 64 + dh] = f2bf(v);
            }
        }
}

// =====================================================================
// Fused residual + LayerNorm: xo = LN(x+y)*g+b (f32), xb = bf16(xo)
// =====================================================================
__global__ __launch_bounds__(256) void addln_kernel(const float* __restrict__ x,
                                                    const float* __restrict__ y,
                                                    const float* __restrict__ g,
                                                    const float* __restrict__ be,
                                                    float* __restrict__ xo,
                                                    bf16_t* __restrict__ xb) {
    __shared__ float red[256];
    const int row = blockIdx.x;
    const int tid = threadIdx.x;
    const float* xr = x + (size_t)row * DMODEL;
    const float* yr = y + (size_t)row * DMODEL;

    float v[4];
    float s = 0.0f;
#pragma unroll
    for (int j = 0; j < 4; ++j) {
        v[j] = xr[tid + j * 256] + yr[tid + j * 256];
        s += v[j];
    }
    red[tid] = s; __syncthreads();
    for (int off = 128; off > 0; off >>= 1) {
        if (tid < off) red[tid] += red[tid + off];
        __syncthreads();
    }
    float mean = red[0] * (1.0f / DMODEL);
    __syncthreads();

    float ss = 0.0f;
#pragma unroll
    for (int j = 0; j < 4; ++j) { float d = v[j] - mean; ss += d * d; }
    red[tid] = ss; __syncthreads();
    for (int off = 128; off > 0; off >>= 1) {
        if (tid < off) red[tid] += red[tid + off];
        __syncthreads();
    }
    float rstd = rsqrtf(red[0] * (1.0f / DMODEL) + 1e-5f);

#pragma unroll
    for (int j = 0; j < 4; ++j) {
        int idx = tid + j * 256;
        float o = (v[j] - mean) * rstd * g[idx] + be[idx];
        xo[(size_t)row * DMODEL + idx] = o;
        xb[(size_t)row * DMODEL + idx] = f2bf(o);
    }
}

// =====================================================================
// Host orchestration
// =====================================================================
extern "C" void kernel_launch(void* const* d_in, const int* in_sizes, int n_in,
                              void* d_out, int out_size, void* d_ws, size_t ws_size,
                              hipStream_t stream) {
    (void)in_sizes; (void)n_in; (void)out_size; (void)ws_size;

    const float* x_in = (const float*)d_in[0];
    const float* mask = (const float*)d_in[1];
    const float* Wq = (const float*)d_in[2];  const float* bq = (const float*)d_in[3];
    const float* Wk = (const float*)d_in[4];  const float* bk = (const float*)d_in[5];
    const float* Wv = (const float*)d_in[6];  const float* bv = (const float*)d_in[7];
    const float* Wo = (const float*)d_in[8];  const float* bo = (const float*)d_in[9];
    const float* W1 = (const float*)d_in[10]; const float* b1 = (const float*)d_in[11];
    const float* W2 = (const float*)d_in[12]; const float* b2 = (const float*)d_in[13];
    const float* lng = (const float*)d_in[14];
    const float* lnb = (const float*)d_in[15];

    const size_t nBSD = (size_t)BB * SS * DMODEL;      // 4M
    const size_t nBSF = (size_t)BB * SS * DFF;         // 16M
    const int Mrows = BB * SS;                         // 4096

    char* w = (char*)d_ws;
    auto alloc = [&](size_t bytes) -> void* {
        void* p = (void*)w;
        w += (bytes + 255) & ~(size_t)255;
        return p;
    };
    float*  xf  = (float*) alloc(nBSD * sizeof(float));
    bf16_t* xb  = (bf16_t*)alloc(nBSD * sizeof(bf16_t));
    bf16_t* qb  = (bf16_t*)alloc(nBSD * sizeof(bf16_t));
    bf16_t* kb_ = (bf16_t*)alloc(nBSD * sizeof(bf16_t));
    bf16_t* vb_ = (bf16_t*)alloc(nBSD * sizeof(bf16_t));
    bf16_t* ab  = (bf16_t*)alloc(nBSD * sizeof(bf16_t));
    float*  yf  = (float*) alloc(nBSD * sizeof(float));
    bf16_t* hb  = (bf16_t*)alloc(nBSF * sizeof(bf16_t));

    auto gemm = [&](const bf16_t* A, const float* W_, const float* bias,
                    float* Cf, bf16_t* Cb, int N, int K, float scale, int relu) {
        dim3 grid(N / BN, Mrows / BM);
        gemm_kernel<<<grid, 256, 0, stream>>>(A, W_, bias, Cf, Cb, Mrows, N, K, scale, relu);
    };

    posenc_kernel<<<(unsigned)(nBSD / 256), 256, 0, stream>>>(x_in, xf, xb);

    const float qscale = 0.125f;   // DH^-0.5
    for (int i = 0; i < LAY; ++i) {
        const size_t dd = (size_t)DMODEL * DMODEL;
        const size_t df = (size_t)DMODEL * DFF;
        gemm(xb, Wq + i * dd, bq + i * DMODEL, nullptr, qb, DMODEL, DMODEL, qscale, 0);
        gemm(xb, Wk + i * dd, bk + i * DMODEL, nullptr, kb_, DMODEL, DMODEL, 1.0f, 0);
        gemm(xb, Wv + i * dd, bv + i * DMODEL, nullptr, vb_, DMODEL, DMODEL, 1.0f, 0);
        attn_kernel<<<dim3(SS / 128, BB * NH), 128, 0, stream>>>(qb, kb_, vb_, mask, ab);
        gemm(ab, Wo + i * dd, bo + i * DMODEL, yf, nullptr, DMODEL, DMODEL, 1.0f, 0);
        addln_kernel<<<Mrows, 256, 0, stream>>>(xf, yf, lng + (size_t)(2 * i) * DMODEL,
                                                lnb + (size_t)(2 * i) * DMODEL, xf, xb);
        gemm(xb, W1 + i * df, b1 + i * DFF, nullptr, hb, DFF, DMODEL, 1.0f, 1);
        gemm(hb, W2 + i * df, b2 + i * DMODEL, yf, nullptr, DMODEL, DFF, 1.0f, 0);
        addln_kernel<<<Mrows, 256, 0, stream>>>(xf, yf, lng + (size_t)(2 * i + 1) * DMODEL,
                                                lnb + (size_t)(2 * i + 1) * DMODEL, xf, xb);
    }

    (void)hipMemcpyAsync(d_out, xf, nBSD * sizeof(float), hipMemcpyDeviceToDevice, stream);
}
